// FC_37958920962093
// MI455X (gfx1250) — 8-bit matrix operand order repaired and checked by reading, compiled, not run on hardware
//
#include <hip/hip_runtime.h>
#include <hip/hip_bf16.h>
#include <cstdint>

typedef int          v2i  __attribute__((ext_vector_type(2)));
typedef int          v4i  __attribute__((ext_vector_type(4)));
typedef int          v8i  __attribute__((ext_vector_type(8)));
typedef int          v16i __attribute__((ext_vector_type(16)));
typedef unsigned int v4u  __attribute__((ext_vector_type(4)));
typedef float        v8f  __attribute__((ext_vector_type(8)));

#define HDIM  4096
#define BATCH 8192

#define FP8_P1 0x38u
#define FP8_M1 0xB8u

#define LROW 144


__global__ __launch_bounds__(256) void quant_act4(const float4* __restrict__ x,
                                                  unsigned int* __restrict__ q, int n4) {
    int i = blockIdx.x * blockDim.x + threadIdx.x;
    if (i >= n4) return;
    float4 v = x[i];
    unsigned int b0 = (v.x >= 0.5f) ? FP8_P1 : FP8_M1;
    unsigned int b1 = (v.y >= 0.5f) ? FP8_P1 : FP8_M1;
    unsigned int b2 = (v.z >= 0.5f) ? FP8_P1 : FP8_M1;
    unsigned int b3 = (v.w >= 0.5f) ? FP8_P1 : FP8_M1;
    q[i] = b0 | (b1 << 8) | (b2 << 16) | (b3 << 24);
}

__global__ __launch_bounds__(256) void quant_w4(const float4* __restrict__ w,
                                                unsigned int* __restrict__ q, int n4) {
    int i = blockIdx.x * blockDim.x + threadIdx.x;
    if (i >= n4) return;
    float4 v = w[i];
    unsigned int b0 = (v.x >= 0.0f) ? FP8_P1 : FP8_M1;
    unsigned int b1 = (v.y >= 0.0f) ? FP8_P1 : FP8_M1;
    unsigned int b2 = (v.z >= 0.0f) ? FP8_P1 : FP8_M1;
    unsigned int b3 = (v.w >= 0.0f) ? FP8_P1 : FP8_M1;
    q[i] = b0 | (b1 << 8) | (b2 << 16) | (b3 << 24);
}

__global__ __launch_bounds__(256) void quant_wout(const float* __restrict__ w,
                                                  unsigned char* __restrict__ q) {
    int i = blockIdx.x * blockDim.x + threadIdx.x;
    if (i >= 16 * HDIM) return;
    int row = i >> 12;
    q[i] = (row < 10) ? ((w[i] >= 0.0f) ? (unsigned char)FP8_P1 : (unsigned char)FP8_M1)
                      : (unsigned char)0;
}

#if __has_builtin(__builtin_amdgcn_tensor_load_to_lds)
#define HAVE_TDM 1
__device__ __forceinline__ void tdm_load_tile(unsigned lds_off, const unsigned char* gptr,
                                              unsigned rows) {
    unsigned long long ga = (unsigned long long)(uintptr_t)gptr;
    v4u g0;
    g0[0] = 1u;
    g0[1] = lds_off;
    g0[2] = (unsigned)(ga & 0xFFFFFFFFu);
    g0[3] = (unsigned)((ga >> 32) & 0x1FFFFFFu) | (2u << 30);
    v8i g1;
    g1[0] = (int)((1u << 20) | (4u << 22) | (3u << 25));
    g1[1] = (int)((HDIM & 0xFFFFu) << 16);
    g1[2] = (int)((HDIM & 0xFFFFu) << 16);
    g1[3] = (int)(128u << 16);
    g1[4] = (int)rows;
    g1[5] = (int)HDIM;
    g1[6] = 0;
    g1[7] = 0;
    v4i z4 = {0, 0, 0, 0};
    v8i z8 = {0, 0, 0, 0, 0, 0, 0, 0};
    __builtin_amdgcn_tensor_load_to_lds(g0, g1, z4, z4, z8, 0);
}
#else
#define HAVE_TDM 0
#endif

__global__ __launch_bounds__(256)
void bgemm_bn_bin(const unsigned char* __restrict__ A,
                  const unsigned char* __restrict__ Wq,
                  const float* __restrict__ gamma, const float* __restrict__ beta,
                  const float* __restrict__ mean,  const float* __restrict__ var,
                  unsigned char* __restrict__ Out)
{
    __shared__ unsigned char shA[2][128 * LROW];
    __shared__ unsigned char shB[2][256 * LROW];

    const int lane = threadIdx.x & 31;
    const int wave = threadIdx.x >> 5;
    const int wm = wave & 1;
    const int wn = wave >> 1;
    const int blkM = blockIdx.x * 128;
    const int blkN = blockIdx.y * 256;
    const int half = lane >> 4;
    const int lid  = lane & 15;

    const unsigned char* Ablk = A  + (size_t)blkM * HDIM;
    const unsigned char* Bblk = Wq + (size_t)blkN * HDIM;
    const unsigned ldsA = (unsigned)(uintptr_t)&shA[0][0];
    const unsigned ldsB = (unsigned)(uintptr_t)&shB[0][0];

    v8f acc[4][4] = {};

#if HAVE_TDM
    if (wave == 0) {
        tdm_load_tile(ldsA, Ablk, 128);
        tdm_load_tile(ldsB, Bblk, 256);
    }
#endif

    for (int kb = 0; kb < HDIM; kb += 128) {
        const int buf = (kb >> 7) & 1;
        __syncthreads();
#if HAVE_TDM
        if (wave == 0) {
            if (kb + 128 < HDIM) {
                tdm_load_tile(ldsA + (buf ^ 1) * (128 * LROW),
                              Ablk + kb + 128, 128);
                tdm_load_tile(ldsB + (buf ^ 1) * (256 * LROW),
                              Bblk + kb + 128, 256);
                __builtin_amdgcn_s_wait_tensorcnt(2);
            } else {
                __builtin_amdgcn_s_wait_tensorcnt(0);
            }
        }
#else
        for (int c = threadIdx.x; c < 128 * 8; c += 256) {
            int r = c >> 3, o = (c & 7) * 16;
            *(v4i*)(&shA[buf][0] + r * LROW + o) =
                *(const v4i*)(Ablk + (size_t)r * HDIM + kb + o);
        }
        for (int c = threadIdx.x; c < 256 * 8; c += 256) {
            int r = c >> 3, o = (c & 7) * 16;
            *(v4i*)(&shB[buf][0] + r * LROW + o) =
                *(const v4i*)(Bblk + (size_t)r * HDIM + kb + o);
        }
#endif
        __syncthreads();

        v16i bfrag[4];
#pragma unroll
        for (int j = 0; j < 4; ++j) {
            const unsigned char* bp =
                &shB[buf][0] + (wn * 64 + 16 * j + lid) * LROW + 16 * half;
            v4i* b = (v4i*)&bfrag[j];
            b[0] = *(const v4i*)(bp);
            b[1] = *(const v4i*)(bp + 32);
            b[2] = *(const v4i*)(bp + 64);
            b[3] = *(const v4i*)(bp + 96);
        }
#pragma unroll
        for (int i = 0; i < 4; ++i) {
            const unsigned char* ap =
                &shA[buf][0] + (wm * 64 + 16 * i + lid) * LROW + 16 * half;
            v16i afrag;
            v2i* a = (v2i*)&afrag;
#pragma unroll
            for (int p = 0; p < 8; ++p)
                a[p] = *(const v2i*)(ap + 32 * (p >> 1) + 8 * (p & 1));
#pragma unroll
            for (int j = 0; j < 4; ++j)
                acc[i][j] = __builtin_amdgcn_wmma_f32_16x16x128_fp8_fp8(
                    afrag, bfrag[j], (short)0, acc[i][j], false, false);
        }
    }

#pragma unroll
    for (int j = 0; j < 4; ++j) {
        const int col = blkN + wn * 64 + 16 * j + lid;
        const float rs = rsqrtf(var[col] + 1e-5f);
        const float g  = gamma[col] * rs;
        const float mu = mean[col];
        const float bt = beta[col];
#pragma unroll
        for (int i = 0; i < 4; ++i) {
#pragma unroll
            for (int r = 0; r < 8; ++r) {
                const int row = blkM + wm * 64 + 16 * i + 8 * half + r;
                const float y = (acc[i][j][r] - mu) * g + bt;
                Out[(size_t)row * HDIM + col] =
                    (y >= 0.0f) ? (unsigned char)FP8_P1 : (unsigned char)FP8_M1;
            }
        }
    }
}

__global__ __launch_bounds__(256)
void head_gemm_tn(const unsigned char* __restrict__ A,
                  const unsigned char* __restrict__ Wq,
                  const float* __restrict__ tn_w, const float* __restrict__ tn_b,
                  const float* __restrict__ tn_m, const float* __restrict__ tn_v,
                  float* __restrict__ out)
{
    const int lane = threadIdx.x & 31;
    const int wave = threadIdx.x >> 5;
    const int m0 = blockIdx.x * 512 + wave * 64;
    const int half = lane >> 4;
    const int lid  = lane & 15;

    v8f acc[4] = {};

    for (int kb = 0; kb < HDIM; kb += 128) {
        v16i bfrag;
        {
            const unsigned char* bp = Wq + (size_t)lid * HDIM + kb + 16 * half;
            v4i* b = (v4i*)&bfrag;
            b[0] = *(const v4i*)(bp);
            b[1] = *(const v4i*)(bp + 32);
            b[2] = *(const v4i*)(bp + 64);
            b[3] = *(const v4i*)(bp + 96);
        }
#pragma unroll
        for (int i = 0; i < 4; ++i) {
            const unsigned char* ap =
                A + (size_t)(m0 + 16 * i + lid) * HDIM + kb + 16 * half;
            v16i afrag;
            v2i* a = (v2i*)&afrag;
#pragma unroll
            for (int p = 0; p < 8; ++p)
                a[p] = *(const v2i*)(ap + 32 * (p >> 1) + 8 * (p & 1));
            acc[i] = __builtin_amdgcn_wmma_f32_16x16x128_fp8_fp8(
                afrag, bfrag, (short)0, acc[i], false, false);
        }
    }

    const float scale = tn_w[0] * rsqrtf(tn_v[0] + 1e-4f);
    const float mu = tn_m[0];
    const float bt = tn_b[0];
    const int col = lid;
    if (col < 10) {
#pragma unroll
        for (int i = 0; i < 4; ++i) {
#pragma unroll
            for (int r = 0; r < 8; ++r) {
                const int row = m0 + 16 * i + 8 * half + r;
                out[(size_t)row * 10 + col] = (acc[i][r] - mu) * scale + bt;
            }
        }
    }
}

extern "C" void kernel_launch(void* const* d_in, const int* in_sizes, int n_in,
                              void* d_out, int out_size, void* d_ws, size_t ws_size,
                              hipStream_t stream) {
    const float* x        = (const float*)d_in[0];
    const float* W        = (const float*)d_in[1];
    const float* Wout     = (const float*)d_in[2];
    const float* bn_gamma = (const float*)d_in[3];
    const float* bn_beta  = (const float*)d_in[4];
    const float* bn_mean  = (const float*)d_in[5];
    const float* bn_var   = (const float*)d_in[6];
    const float* tn_w     = (const float*)d_in[7];
    const float* tn_b     = (const float*)d_in[8];
    const float* tn_m     = (const float*)d_in[9];
    const float* tn_v     = (const float*)d_in[10];

    const size_t actBytes = (size_t)BATCH * HDIM;
    const size_t wBytes   = (size_t)HDIM * HDIM;
    uint8_t* ws    = (uint8_t*)d_ws;
    unsigned char* actA  = ws;
    unsigned char* actB  = ws + actBytes;
    unsigned char* wq    = ws + 2 * actBytes;
    unsigned char* woutq = ws + 2 * actBytes + wBytes;

    {
        int n4 = (int)((size_t)BATCH * HDIM / 4);
        quant_act4<<<(n4 + 255) / 256, 256, 0, stream>>>(
            (const float4*)x, (unsigned int*)actA, n4);
    }

    unsigned char* in   = actA;
    unsigned char* outb = actB;
    for (int l = 0; l < 3; ++l) {
        int n4 = (int)((size_t)HDIM * HDIM / 4);
        quant_w4<<<(n4 + 255) / 256, 256, 0, stream>>>(
            (const float4*)(W + (size_t)l * HDIM * HDIM), (unsigned int*)wq, n4);
        dim3 grid(BATCH / 128, HDIM / 256);
        bgemm_bn_bin<<<grid, 256, 0, stream>>>(
            in, wq,
            bn_gamma + (size_t)l * HDIM, bn_beta + (size_t)l * HDIM,
            bn_mean  + (size_t)l * HDIM, bn_var  + (size_t)l * HDIM,
            outb);
        unsigned char* t = in; in = outb; outb = t;
    }

    quant_wout<<<(16 * HDIM + 255) / 256, 256, 0, stream>>>(Wout, woutq);
    head_gemm_tn<<<BATCH / 512, 256, 0, stream>>>(
        in, woutq, tn_w, tn_b, tn_m, tn_v, (float*)d_out);
}
